// DPMultiheadAttention_77704548319557
// MI455X (gfx1250) — compile-verified, matrix operand order checked by reading, not run on hardware
//
#include <hip/hip_runtime.h>
#include <hip/hip_bf16.h>

typedef __attribute__((ext_vector_type(16))) __bf16 v16bf;
typedef __attribute__((ext_vector_type(8)))  float  v8f;

#define E_DIM   1024
#define L_SEQ   2048
#define N_BATCH 2
#define N_HEADS 16
#define D_HEAD  64
#define M_ROWS  (L_SEQ * N_BATCH)   // 4096

// ---------------------------------------------------------------------------
// gfx1250 async-to-LDS (tracked by ASYNCcnt).  GVS addressing:
//   lds[vdst + ioffset] = mem[saddr + vaddr + ioffset]   (16 bytes)
// ---------------------------------------------------------------------------
__device__ __forceinline__ void async_b128_to_lds(const void* gbase,
                                                  unsigned voff_bytes,
                                                  const void* lds_dst) {
  const unsigned long long base = (unsigned long long)(uintptr_t)gbase;
  const unsigned lds = (unsigned)(uintptr_t)lds_dst;   // low 32b = LDS offset
  asm volatile("global_load_async_to_lds_b128 %0, %1, %2 offset:0"
               :: "v"(lds), "v"(voff_bytes), "s"(base)
               : "memory");
}

#if defined(__has_builtin)
#  if __has_builtin(__builtin_amdgcn_s_wait_asynccnt)
#    define WAIT_ASYNC(n) __builtin_amdgcn_s_wait_asynccnt(n)
#  endif
#endif
#ifndef WAIT_ASYNC
#  define WAIT_ASYNC(n) asm volatile("s_wait_asynccnt %0" :: "i"(n) : "memory")
#endif

// ---------------------------------------------------------------------------
// WMMA helpers (CDNA5 gfx1250, wave32)
// ---------------------------------------------------------------------------
__device__ __forceinline__ v8f wmma_bf16(v16bf a, v16bf b, v8f c) {
  // D(f32 16x16) = A(bf16 16x32) x B(bf16 32x16) + C
  return __builtin_amdgcn_wmma_f32_16x16x32_bf16(
      /*neg_a=*/false, a, /*neg_b=*/false, b,
      /*c_mod=*/(short)0, c, /*reuse_a=*/false, /*reuse_b=*/false);
}

// 16x32 bf16 fragment from a row-major tile.  ISA layout: lane&15 = row,
// lane>>4 selects K-halves {0..7,16..23} / {8..15,24..31}.  Two 16B loads.
__device__ __forceinline__ v16bf frag_load(const __bf16* base, int stride, int lane) {
  const int row = lane & 15;
  const int kb  = (lane >> 4) << 3;
  const __bf16* p = base + row * stride + kb;
  v16bf a;
  ((uint4*)&a)[0] = *(const uint4*)(p);        // K = kb+0 .. kb+7
  ((uint4*)&a)[1] = *(const uint4*)(p + 16);   // K = kb+16 .. kb+23
  return a;
}

// ---------------------------------------------------------------------------
// Stage 1: QKV projection.  out[m,f] = sum_e X[m,e]*W[f,e] (+bias, q scaled)
// 256x64 tile per block, 8 waves x (32 rows x 64 cols) = 8 WMMA / k-step.
// All fragments preloaded per k-step so ds_loads overlap the WMMA pipe.
// Q,K stored bf16 (B,H,L,D); V stored transposed bf16 (B,H,D,L).
// ---------------------------------------------------------------------------
__global__ __launch_bounds__(256) void qkv_kernel(
    const float* __restrict__ query, const float* __restrict__ key,
    const float* __restrict__ value, const float* __restrict__ w_in,
    const float* __restrict__ b_in,
    __bf16* __restrict__ Qb, __bf16* __restrict__ Kb, __bf16* __restrict__ Vt) {
  constexpr int TM = 256, TN = 64, TK = 32;
  constexpr int SA = TK + 8;
  constexpr int SB = TK + 8;
  __shared__ __bf16 lA[TM * SA];
  __shared__ __bf16 lB[TN * SB];

  const int s = blockIdx.z;                    // 0=q 1=k 2=v
  const float* X    = (s == 0) ? query : (s == 1) ? key : value;
  const float* W    = w_in + (size_t)s * E_DIM * E_DIM;
  const float* bias = b_in + s * E_DIM;

  const int n0   = blockIdx.x * TN;
  const int m0   = blockIdx.y * TM;
  const int t    = threadIdx.x;
  const int lane = t & 31;
  const int wv   = t >> 5;
  const int wrow = wv * 32;                    // 32 rows per wave

  v8f acc[2][4] = {};

  for (int k0 = 0; k0 < E_DIM; k0 += TK) {
    // prefetch next K-tile (global_prefetch_b8)
    if (k0 + TK < E_DIM) {
      __builtin_prefetch(X + (size_t)(m0 + t) * E_DIM + k0 + TK, 0, 1);
      __builtin_prefetch(W + (size_t)(n0 + (t & 63)) * E_DIM + k0 + TK, 0, 1);
    }
    // A tile: 256x32 fp32 -> bf16 (2048 float4 / 256 thr = 8 each)
#pragma unroll
    for (int i = 0; i < 8; ++i) {
      const int idx = i * 256 + t;
      const int row = idx >> 3;
      const int kq  = (idx & 7) * 4;
      const float4 f = *(const float4*)(X + (size_t)(m0 + row) * E_DIM + k0 + kq);
      __bf16* dst = &lA[row * SA + kq];
      dst[0] = (__bf16)f.x; dst[1] = (__bf16)f.y;
      dst[2] = (__bf16)f.z; dst[3] = (__bf16)f.w;
    }
    // B tile: 64x32 fp32 -> bf16
#pragma unroll
    for (int i = 0; i < 2; ++i) {
      const int idx = i * 256 + t;
      const int row = idx >> 3;
      const int kq  = (idx & 7) * 4;
      const float4 f = *(const float4*)(W + (size_t)(n0 + row) * E_DIM + k0 + kq);
      __bf16* dst = &lB[row * SB + kq];
      dst[0] = (__bf16)f.x; dst[1] = (__bf16)f.y;
      dst[2] = (__bf16)f.z; dst[3] = (__bf16)f.w;
    }
    __syncthreads();
    // preload ALL fragments of this k-step, then back-to-back WMMAs
    const v16bf a0 = frag_load(&lA[wrow * SA], SA, lane);
    const v16bf a1 = frag_load(&lA[(wrow + 16) * SA], SA, lane);
    v16bf bf[4];
#pragma unroll
    for (int nt = 0; nt < 4; ++nt) bf[nt] = frag_load(&lB[nt * 16 * SB], SB, lane);
#pragma unroll
    for (int nt = 0; nt < 4; ++nt) {
      acc[0][nt] = wmma_bf16(a0, bf[nt], acc[0][nt]);
      acc[1][nt] = wmma_bf16(a1, bf[nt], acc[1][nt]);
    }
    __syncthreads();
  }

  const float qscale = (s == 0) ? 0.125f : 1.0f;   // 1/sqrt(64)
  const int rowoff = (lane >> 4) * 8;
#pragma unroll
  for (int g = 0; g < 2; ++g) {
#pragma unroll
    for (int nt = 0; nt < 4; ++nt) {
      const int n  = n0 + nt * 16 + (lane & 15);
      const float bv = bias[n];
      const int h = n >> 6;
      const int d = n & 63;
#pragma unroll
      for (int r = 0; r < 8; ++r) {
        const int m  = m0 + wrow + g * 16 + rowoff + r;
        const int l  = m >> 1;                 // m = l*B + b, B=2
        const int bb = m & 1;
        const __bf16 o = (__bf16)((acc[g][nt][r] + bv) * qscale);
        if (s == 2) {
          Vt[(((size_t)bb * N_HEADS + h) * D_HEAD + d) * L_SEQ + l] = o;
        } else {
          __bf16* dstp = (s == 0) ? Qb : Kb;
          dstp[(((size_t)bb * N_HEADS + h) * L_SEQ + l) * D_HEAD + d] = o;
        }
      }
    }
  }
}

// ---------------------------------------------------------------------------
// Stage 2: flash attention.  grid = (L/128, B*H); 8 waves x 16 q-rows.
// K/V tiles staged once per block via double-buffered async-to-LDS DMA;
// scores/softmax stay in wave-private LDS slices (no HBM traffic).
// ---------------------------------------------------------------------------
__global__ __launch_bounds__(256) void attn_kernel(
    const __bf16* __restrict__ Qb, const __bf16* __restrict__ Kb,
    const __bf16* __restrict__ Vt, __bf16* __restrict__ Ctx) {
  __shared__ __bf16 sK[2][32 * D_HEAD];   // 32 keys x 64 d   (4 KB each)
  __shared__ __bf16 sV[2][D_HEAD * 32];   // 64 d   x 32 keys (4 KB each)
  __shared__ float  sS[8][16 * 32];       // raw scores (f32)
  __shared__ __bf16 sP[8][16 * 32];       // probabilities (bf16)
  __shared__ float  sFac[8][16];          // per-row broadcast

  const int t    = threadIdx.x;
  const int lane = t & 31;
  const int wv   = t >> 5;
  const int bh   = blockIdx.y;
  const int bb   = bh / N_HEADS;
  const int h    = bh % N_HEADS;
  const __bf16* Q = Qb + (size_t)bh * L_SEQ * D_HEAD;
  const __bf16* K = Kb + (size_t)bh * L_SEQ * D_HEAD;
  const __bf16* V = Vt + (size_t)bh * D_HEAD * L_SEQ;

  const int q0 = blockIdx.x * 128 + wv * 16;

  // Q fragments: 16x64 = two 16x32 A-frags held in registers throughout.
  const v16bf qa0 = frag_load(Q + (size_t)q0 * D_HEAD +  0, D_HEAD, lane);
  const v16bf qa1 = frag_load(Q + (size_t)q0 * D_HEAD + 32, D_HEAD, lane);

  // Issue one K+V tile into LDS buffer `buf` (2 async b128 ops per thread).
  auto issue_tile = [&](int kb0, int buf) {
    // K tile: 32 rows x 64 bf16 = one contiguous 4KB chunk.
    async_b128_to_lds(K + (size_t)kb0 * D_HEAD, (unsigned)(t * 16),
                      &sK[buf][t * 8]);
    // V tile: 64 rows (d) x 32 keys, global row stride L*2 bytes.
    const int row = t >> 2, ch = t & 3;
    async_b128_to_lds(V + kb0,
                      (unsigned)(row * (L_SEQ * 2) + ch * 16),
                      &sV[buf][row * 32 + ch * 8]);
  };

  v8f acc[4] = {};
  const int srow = lane & 15;
  const int half = lane >> 4;
  float m_run = -3.0e38f;
  float l_run = 0.0f;

  constexpr int NT = L_SEQ / 32;
  issue_tile(0, 0);

  for (int it = 0; it < NT; ++it) {
    const int cur = it & 1;
    if (it + 1 < NT) {
      issue_tile((it + 1) * 32, cur ^ 1);
      WAIT_ASYNC(2);       // loads retire in order -> tile `it` complete
    } else {
      WAIT_ASYNC(0);
    }
    __syncthreads();

    // ---- scores: 16 q-rows x 32 keys (preload all K-frags, then WMMA) ----
    v16bf kf[2][2];
#pragma unroll
    for (int nt = 0; nt < 2; ++nt) {
      kf[nt][0] = frag_load(&sK[cur][(nt * 16) * D_HEAD +  0], D_HEAD, lane);
      kf[nt][1] = frag_load(&sK[cur][(nt * 16) * D_HEAD + 32], D_HEAD, lane);
    }
#pragma unroll
    for (int nt = 0; nt < 2; ++nt) {
      v8f c = {};
      c = wmma_bf16(qa0, kf[nt][0], c);
      c = wmma_bf16(qa1, kf[nt][1], c);
      const int col = nt * 16 + (lane & 15);
      const int ro  = (lane >> 4) * 8;
#pragma unroll
      for (int r = 0; r < 8; ++r) sS[wv][(ro + r) * 32 + col] = c[r];
    }

    // ---- online softmax (wave-private, lockstep; no block barrier) ----
    const float* Sr = &sS[wv][srow * 32 + half * 16];
    float mx = Sr[0];
#pragma unroll
    for (int j = 1; j < 16; ++j) mx = fmaxf(mx, Sr[j]);
    mx = fmaxf(mx, __shfl_xor(mx, 16, 32));
    const float newm = fmaxf(m_run, mx);
    const float fac  = __expf(m_run - newm);
    float sum = 0.0f;
    __bf16* Pr = &sP[wv][srow * 32 + half * 16];
#pragma unroll
    for (int j = 0; j < 16; ++j) {
      const float p = __expf(Sr[j] - newm);
      sum += p;
      Pr[j] = (__bf16)p;
    }
    sum += __shfl_xor(sum, 16, 32);
    l_run = l_run * fac + sum;
    m_run = newm;
    if (lane < 16) sFac[wv][srow] = fac;

    float frs[8];
#pragma unroll
    for (int r = 0; r < 8; ++r)
      frs[r] = (lane < 16) ? sFac[wv][r] : sFac[wv][r + 8];
#pragma unroll
    for (int dt = 0; dt < 4; ++dt)
#pragma unroll
      for (int r = 0; r < 8; ++r) acc[dt][r] *= frs[r];

    // ---- PV: acc(16x64) += P(16x32) x V(32x64), V-frags preloaded ----
    v16bf vf[4];
#pragma unroll
    for (int dt = 0; dt < 4; ++dt)
      vf[dt] = frag_load(&sV[cur][(dt * 16) * 32], 32, lane);
    const v16bf pa = frag_load(&sP[wv][0], 32, lane);
#pragma unroll
    for (int dt = 0; dt < 4; ++dt)
      acc[dt] = wmma_bf16(pa, vf[dt], acc[dt]);
    __syncthreads();   // all waves done with buffer `cur` before re-fill
  }

  // ---- normalize by 1/l and store ctx bf16 (M x E) ----
  if (lane < 16) sFac[wv][srow] = 1.0f / l_run;
  float inv8[8];
#pragma unroll
  for (int r = 0; r < 8; ++r)
    inv8[r] = (lane < 16) ? sFac[wv][r] : sFac[wv][r + 8];
#pragma unroll
  for (int dt = 0; dt < 4; ++dt) {
    const int col = h * D_HEAD + dt * 16 + (lane & 15);
#pragma unroll
    for (int r = 0; r < 8; ++r) {
      const int lrow = q0 + ((lane < 16) ? r : r + 8);
      Ctx[((size_t)lrow * N_BATCH + bb) * E_DIM + col] = (__bf16)(acc[dt][r] * inv8[r]);
    }
  }
}

// ---------------------------------------------------------------------------
// Stage 3: output projection.  out[m,n] = sum_e ctx[m,e]*w_out[n,e] + b_out[n]
// 256x64 tile, A tile (bf16) DMA'd via async-to-LDS; B converted on VALU.
// ---------------------------------------------------------------------------
__global__ __launch_bounds__(256) void out_kernel(
    const __bf16* __restrict__ Ctx, const float* __restrict__ w_out,
    const float* __restrict__ b_out, float* __restrict__ Out) {
  constexpr int TM = 256, TN = 64, TK = 32;
  constexpr int SA = TK + 8;
  constexpr int SB = TK + 8;
  __shared__ __bf16 lA[TM * SA];
  __shared__ __bf16 lB[TN * SB];

  const int n0   = blockIdx.x * TN;
  const int m0   = blockIdx.y * TM;
  const int t    = threadIdx.x;
  const int lane = t & 31;
  const int wv   = t >> 5;
  const int wrow = wv * 32;

  v8f acc[2][4] = {};

  for (int k0 = 0; k0 < E_DIM; k0 += TK) {
    // A tile: 256x32 bf16, async DMA (4 x b128 per thread)
#pragma unroll
    for (int i = 0; i < 4; ++i) {
      const int idx = i * 256 + t;
      const int row = idx >> 2;
      const int kq  = (idx & 3) * 8;
      async_b128_to_lds(Ctx + (size_t)m0 * E_DIM + k0,
                        (unsigned)(row * (E_DIM * 2) + kq * 2),
                        &lA[row * SA + kq]);
    }
    // B tile: 64x32 fp32 -> bf16
#pragma unroll
    for (int i = 0; i < 2; ++i) {
      const int idx = i * 256 + t;
      const int row = idx >> 3;
      const int kq  = (idx & 7) * 4;
      const float4 f = *(const float4*)(w_out + (size_t)(n0 + row) * E_DIM + k0 + kq);
      __bf16* dst = &lB[row * SB + kq];
      dst[0] = (__bf16)f.x; dst[1] = (__bf16)f.y;
      dst[2] = (__bf16)f.z; dst[3] = (__bf16)f.w;
    }
    WAIT_ASYNC(0);
    __syncthreads();
    const v16bf a0 = frag_load(&lA[wrow * SA], SA, lane);
    const v16bf a1 = frag_load(&lA[(wrow + 16) * SA], SA, lane);
    v16bf bf[4];
#pragma unroll
    for (int nt = 0; nt < 4; ++nt) bf[nt] = frag_load(&lB[nt * 16 * SB], SB, lane);
#pragma unroll
    for (int nt = 0; nt < 4; ++nt) {
      acc[0][nt] = wmma_bf16(a0, bf[nt], acc[0][nt]);
      acc[1][nt] = wmma_bf16(a1, bf[nt], acc[1][nt]);
    }
    __syncthreads();
  }

  const int rowoff = (lane >> 4) * 8;
#pragma unroll
  for (int g = 0; g < 2; ++g) {
#pragma unroll
    for (int nt = 0; nt < 4; ++nt) {
      const int n   = n0 + nt * 16 + (lane & 15);
      const float bv = b_out[n];
#pragma unroll
      for (int r = 0; r < 8; ++r) {
        const int m = m0 + wrow + g * 16 + rowoff + r;
        Out[(size_t)m * E_DIM + n] = acc[g][nt][r] + bv;
      }
    }
  }
}

// ---------------------------------------------------------------------------
extern "C" void kernel_launch(void* const* d_in, const int* in_sizes, int n_in,
                              void* d_out, int out_size, void* d_ws, size_t ws_size,
                              hipStream_t stream) {
  const float* query = (const float*)d_in[0];
  const float* key   = (const float*)d_in[1];
  const float* value = (const float*)d_in[2];
  const float* w_in  = (const float*)d_in[3];
  const float* b_in  = (const float*)d_in[4];
  const float* w_out = (const float*)d_in[5];
  const float* b_out = (const float*)d_in[6];
  float* out = (float*)d_out;

  const size_t per = (size_t)N_BATCH * N_HEADS * L_SEQ * D_HEAD;  // 4 Mi elems
  __bf16* Qb  = (__bf16*)d_ws;
  __bf16* Kb  = Qb + per;
  __bf16* Vt  = Kb + per;
  __bf16* Ctx = Vt + per;   // (M x E) bf16

  dim3 blk(256);
  qkv_kernel<<<dim3(E_DIM / 64, M_ROWS / 256, 3), blk, 0, stream>>>(
      query, key, value, w_in, b_in, Qb, Kb, Vt);
  attn_kernel<<<dim3(L_SEQ / 128, N_BATCH * N_HEADS), blk, 0, stream>>>(
      Qb, Kb, Vt, Ctx);
  out_kernel<<<dim3(E_DIM / 64, M_ROWS / 256), blk, 0, stream>>>(
      Ctx, w_out, b_out, out);
}
